// RateAdaptionEncoder_36326833389748
// MI455X (gfx1250) — matrix operand order repaired and checked by reading, not compiled, not run on hardware
//
#include <hip/hip_runtime.h>

typedef __attribute__((ext_vector_type(16))) __bf16   v16bf;
typedef __attribute__((ext_vector_type(8)))  float    v8f;
typedef __attribute__((ext_vector_type(4)))  unsigned u32x4;

struct FragPair { u32x4 a, b; };   // 32 bytes == one v16bf fragment

#define B_    8
#define C_    256
#define S_    2048
#define NTOK  (B_ * S_)            // 16384 tokens
#define MAXR  256
#define NEXP  8
#define MTILE 64                   // tokens per block (4 WMMA M-tiles)
#define KBN   8                    // K blocks of 32 (C_ / 32)

// Swizzled B-fragment plane: [expert][kbi][ntile][lane][16 bf16], contiguous per lane
#define KB_STRIDE   (16 * 32 * 16)                  // ushorts per kbi block
#define FRAG_ELEMS  (NEXP * KBN * KB_STRIDE)        // 524288 ushorts = 1 MB

// Workspace layout (bytes)
#define OFF_PERM 256
#define OFF_WHI  (OFF_PERM + NEXP * NTOK * 4)
#define OFF_WLO  (OFF_WHI + FRAG_ELEMS * 2)

static __device__ inline v16bf load_b_frag(const unsigned short* p) {
    FragPair s;
    s.a = *(const u32x4*)p;            // 16 bytes
    s.b = *(const u32x4*)(p + 8);      // next 16 bytes (contiguous)
    return __builtin_bit_cast(v16bf, s);
}

static __device__ inline v16bf load_a_frag(const unsigned short* p) {
    FragPair s;
    s.a = *(const u32x4*)p;            // K run [k, k+8)
    s.b = *(const u32x4*)(p + 8);      // K run [k+8, k+16)
    return __builtin_bit_cast(v16bf, s);
}

__global__ void zero_counts_kernel(int* __restrict__ counts) {
    if (threadIdx.x < NEXP) counts[threadIdx.x] = 0;
}

// Split fp32 weight into bf16 hi/lo planes, written directly in WMMA
// B-fragment order: lane q-th element of block (e,kbi,ntile) is
// W[e][kbi*32 + (lane>>4)*16 + q][ntile*16 + (lane&15)] -> contiguous per lane.
// One thread per packed dword (two bf16).
__global__ void wsplit_kernel(const float* __restrict__ w,
                              unsigned* __restrict__ whi,
                              unsigned* __restrict__ wlo) {
    int i = blockIdx.x * blockDim.x + threadIdx.x;   // < FRAG_ELEMS/2 = 262144
    int qd    = i & 7;                 // dword within lane -> q = 2*qd, 2*qd+1
    int lane  = (i >> 3) & 31;
    int ntile = (i >> 8) & 15;
    int kbi   = (i >> 12) & 7;
    int e     = i >> 15;
    int c = kbi * 32 + ((lane >> 4) << 4) + (qd << 1);
    int n = ntile * 16 + (lane & 15);
    float f0 = w[(e * C_ + c)     * MAXR + n];
    float f1 = w[(e * C_ + c + 1) * MAXR + n];
    unsigned u0 = __builtin_bit_cast(unsigned, f0);
    unsigned u1 = __builtin_bit_cast(unsigned, f1);
    whi[i] = (u0 >> 16) | (u1 & 0xFFFF0000u);
    float h0 = __builtin_bit_cast(float, u0 & 0xFFFF0000u);
    float h1 = __builtin_bit_cast(float, u1 & 0xFFFF0000u);
    wlo[i] = (__builtin_bit_cast(unsigned, f0 - h0) >> 16) |
             (__builtin_bit_cast(unsigned, f1 - h1) & 0xFFFF0000u);
}

// Bucket tokens by expert via atomics
__global__ void bucket_kernel(const long long* __restrict__ idx,
                              int* __restrict__ counts,
                              int* __restrict__ perm) {
    int t = blockIdx.x * blockDim.x + threadIdx.x;
    if (t >= NTOK) return;
    int e = (int)idx[t];
    int pos = atomicAdd(&counts[e], 1);
    perm[e * NTOK + pos] = t;
}

// Copy int64 indexes to the tail of d_out (third tuple element)
__global__ void idxcopy_kernel(const long long* __restrict__ idx,
                               long long* __restrict__ dst) {
    int t = blockIdx.x * blockDim.x + threadIdx.x;
    if (t < NTOK) dst[t] = idx[t];
}

// ---------------------------------------------------------------------------
// Grouped GEMM: block = (expert, 64-token tile). 512 threads = 16 waves.
// Wave w owns output columns [w*16, w*16+16). Per K-block it loads one
// pre-swizzled B hi/lo fragment pair (4x b128) and reuses it across 4 M-tiles:
// 12 v_wmma_f32_16x16x32_bf16 per K-step, 96 per wave. Split-bf16 gives
// near-fp32 accuracy. Residual + bias + rate mask fused into the store.
// ---------------------------------------------------------------------------
__global__ void __launch_bounds__(512)
expert_gemm_kernel(const float* __restrict__ x,
                   const float* __restrict__ bias,
                   const int* __restrict__ rate_choice,
                   const int* __restrict__ counts,
                   const int* __restrict__ perm,
                   const unsigned short* __restrict__ whi,
                   const unsigned short* __restrict__ wlo,
                   float* __restrict__ out_x,
                   float* __restrict__ out_mask) {
    const int e    = blockIdx.y;
    const int tile = blockIdx.x;
    const int cnt  = counts[e];
    const int t0   = tile * MTILE;
    if (t0 >= cnt) return;                   // uniform exit: capture-safe grid

    __shared__ unsigned short sAhi[MTILE * C_];   // 32 KB
    __shared__ unsigned short sAlo[MTILE * C_];   // 32 KB
    __shared__ int sTok[MTILE];

    const int tid = threadIdx.x;
    if (tid < MTILE) {
        int tt = t0 + tid;
        sTok[tid] = (tt < cnt) ? perm[e * NTOK + tt] : -1;
    }
    __syncthreads();

    // Stage 64 token rows of x (K=256) into LDS as split bf16 hi/lo
    for (int i = tid; i < MTILE * C_; i += 512) {
        int m = i >> 8, c = i & 255;
        int tok = sTok[m];
        float f = 0.0f;
        if (tok >= 0) {
            int b = tok >> 11, s = tok & 2047;
            f = x[(b * C_ + c) * S_ + s];
        }
        unsigned u = __builtin_bit_cast(unsigned, f);
        sAhi[i] = (unsigned short)(u >> 16);
        float hif = __builtin_bit_cast(float, u & 0xFFFF0000u);
        sAlo[i] = (unsigned short)(__builtin_bit_cast(unsigned, f - hif) >> 16);
    }
    __syncthreads();

    const int wave = tid >> 5;               // 0..15 : N-tile
    const int lane = tid & 31;
    const int half = lane >> 4;
    const int mrow = lane & 15;
    const int ncol = wave * 16 + (lane & 15);

    const size_t bbase = (size_t)(((e * KBN) * 16 + wave) * 32 + lane) * 16;

    v8f acc[4] = {{}, {}, {}, {}};

    for (int kbi = 0; kbi < KBN; ++kbi) {
        const unsigned short* pbh = whi + bbase + (size_t)kbi * KB_STRIDE;
        const unsigned short* pbl = wlo + bbase + (size_t)kbi * KB_STRIDE;
        v16bf bhi = load_b_frag(pbh);
        v16bf blo = load_b_frag(pbl);
        const int abase = kbi * 32 + (half << 4);
#pragma unroll
        for (int mt = 0; mt < 4; ++mt) {
            const int row = (mt * 16 + mrow) * C_ + abase;
            v16bf ahi = load_a_frag(&sAhi[row]);
            v16bf alo = load_a_frag(&sAlo[row]);
            acc[mt] = __builtin_amdgcn_wmma_f32_16x16x32_bf16(
                false, ahi, false, bhi, (short)0, acc[mt], false, false);
            acc[mt] = __builtin_amdgcn_wmma_f32_16x16x32_bf16(
                false, ahi, false, blo, (short)0, acc[mt], false, false);
            acc[mt] = __builtin_amdgcn_wmma_f32_16x16x32_bf16(
                false, alo, false, bhi, (short)0, acc[mt], false, false);
        }
    }

    const int rc = rate_choice[e];
    const float maskv = (ncol < rc) ? 1.0f : 0.0f;
    const float bv = bias[e * MAXR + ncol];

    // C/D layout: VGPR v of tile mt holds (M = mt*16 + v + 8*half, N = lane&15)
#pragma unroll
    for (int mt = 0; mt < 4; ++mt) {
#pragma unroll
        for (int v = 0; v < 8; ++v) {
            int m = mt * 16 + v + (half << 3);
            int tok = sTok[m];
            if (tok < 0) continue;
            int b = tok >> 11, s = tok & 2047;
            float resid = x[(b * C_ + ncol) * S_ + s];
            float val = (acc[mt][v] + resid + bv) * maskv;
            size_t o = ((size_t)(b * MAXR + ncol)) * S_ + s;
            out_x[o] = val;
            out_mask[o] = maskv;
        }
    }
}

extern "C" void kernel_launch(void* const* d_in, const int* in_sizes, int n_in,
                              void* d_out, int out_size, void* d_ws, size_t ws_size,
                              hipStream_t stream) {
    const float*     x    = (const float*)d_in[0];
    const long long* idx  = (const long long*)d_in[1];   // int64 indexes
    const float*     w    = (const float*)d_in[2];
    const float*     bias = (const float*)d_in[3];
    const int*       rc   = (const int*)d_in[4];

    float* out_x    = (float*)d_out;
    float* out_mask = out_x + (size_t)B_ * MAXR * S_;
    long long* out_idx = (long long*)((char*)d_out + (size_t)2 * B_ * MAXR * S_ * sizeof(float));

    char* ws = (char*)d_ws;
    int*            counts = (int*)ws;
    int*            perm   = (int*)(ws + OFF_PERM);
    unsigned short* whi    = (unsigned short*)(ws + OFF_WHI);
    unsigned short* wlo    = (unsigned short*)(ws + OFF_WLO);

    zero_counts_kernel<<<1, 32, 0, stream>>>(counts);
    wsplit_kernel<<<(FRAG_ELEMS / 2) / 256, 256, 0, stream>>>(
        w, (unsigned*)whi, (unsigned*)wlo);
    bucket_kernel<<<(NTOK + 255) / 256, 256, 0, stream>>>(idx, counts, perm);

    dim3 grid(NTOK / MTILE, NEXP);           // fixed grid; early-exit past bucket end
    expert_gemm_kernel<<<grid, 512, 0, stream>>>(x, bias, rc, counts, perm,
                                                 whi, wlo, out_x, out_mask);

    idxcopy_kernel<<<(NTOK + 255) / 256, 256, 0, stream>>>(idx, out_idx);
}
